// LidarGcnLstmNet_51273319579912
// MI455X (gfx1250) — matrix operand order repaired and checked by reading, not compiled, not run on hardware
//
#include <hip/hip_runtime.h>

#define N_NODES 34560
#define N_EDGES 276480
#define N_TOT   (N_EDGES + N_NODES)   // edges + self loops
#define NGRAPH  32
#define NPGRAPH 1080
#define TSTEPS  8
#define DIN 3
#define D1  256
#define D2  128
#define D3  64
#define D4  32
#define LH  128

typedef __attribute__((ext_vector_type(16))) __bf16         bf16x16;
typedef __attribute__((ext_vector_type(8)))  float          f32x8;
typedef __attribute__((ext_vector_type(16))) unsigned short u16x16;
typedef __attribute__((ext_vector_type(8)))  unsigned short u16x8;
typedef __attribute__((ext_vector_type(4)))  unsigned int   u32x4;
typedef __attribute__((ext_vector_type(8)))  int            i32x8;
typedef __attribute__((ext_vector_type(4)))  int            i32x4;

__device__ __forceinline__ unsigned short f32_to_bf16(float f) {
  unsigned u = __float_as_uint(f);
  u += 0x7FFFu + ((u >> 16) & 1u);          // round-to-nearest-even
  return (unsigned short)(u >> 16);
}

// ---------------------------------------------------------------- utilities
__global__ void k_fill_f32(float* p, long n, float v) {
  long i = (long)blockIdx.x * blockDim.x + threadIdx.x;
  if (i < n) p[i] = v;
}

// degree: deg pre-filled with 1.0 (self loop), add 1 per incoming edge
__global__ void k_deg(const int* __restrict__ ei, float* deg) {
  int e = blockIdx.x * blockDim.x + threadIdx.x;
  if (e < N_EDGES) atomicAdd(&deg[ei[N_EDGES + e]], 1.0f);
}

__global__ void k_dis(const float* __restrict__ deg, float* dis) {
  int i = blockIdx.x * blockDim.x + threadIdx.x;
  if (i < N_NODES) dis[i] = rsqrtf(deg[i]);
}

__global__ void k_norm(const int* __restrict__ ei, const float* __restrict__ dis,
                       float* norm) {
  int e = blockIdx.x * blockDim.x + threadIdx.x;
  if (e >= N_TOT) return;
  float v;
  if (e < N_EDGES) v = dis[ei[e]] * dis[ei[N_EDGES + e]];
  else { float d = dis[e - N_EDGES]; v = d * d; }
  norm[e] = v;
}

// scatter-add:  agg[dst] += feat[src] * norm   (edges + self loops)
__global__ void k_edge_agg(const int* __restrict__ ei, const float* __restrict__ norm,
                           const float* __restrict__ feat, float* agg, int O) {
  long total = (long)N_TOT * O;
  long i = (long)blockIdx.x * blockDim.x + threadIdx.x;
  if (i >= total) return;
  int e = (int)(i / O);
  int d = (int)(i - (long)e * O);
  int s, t;
  if (e < N_EDGES) { s = ei[e]; t = ei[N_EDGES + e]; }
  else             { s = t = e - N_EDGES; }
  atomicAdd(&agg[(long)t * O + d], feat[(long)s * O + d] * norm[e]);
}

// layer 1 transform: act = bf16(relu(agg[N,3] @ W1[3,256] + b1))
__global__ void k_conv1(const float* __restrict__ agg, const float* __restrict__ W1,
                        const float* __restrict__ b1, unsigned short* act) {
  long i = (long)blockIdx.x * blockDim.x + threadIdx.x;
  if (i >= (long)N_NODES * D1) return;
  int n = (int)(i / D1);
  int d = (int)(i - (long)n * D1);
  const float* a = agg + (long)n * DIN;
  float h = b1[d] + a[0] * W1[d] + a[1] * W1[D1 + d] + a[2] * W1[2 * D1 + d];
  act[i] = f32_to_bf16(fmaxf(h, 0.0f));
}

// weight transpose+convert:  Wt[o,k] = bf16(W[k,o])
__global__ void k_wt(const float* __restrict__ W, unsigned short* Wt, int K, int O) {
  int i = blockIdx.x * blockDim.x + threadIdx.x;
  if (i >= K * O) return;
  int o = i / K;
  int k = i - o * K;
  Wt[i] = f32_to_bf16(W[(long)k * O + o]);
}

// act = bf16(relu(agg + b))
__global__ void k_bias_relu_cvt(const float* __restrict__ agg, const float* __restrict__ b,
                                unsigned short* act, int O, long total) {
  long i = (long)blockIdx.x * blockDim.x + threadIdx.x;
  if (i >= total) return;
  int d = (int)(i % O);
  act[i] = f32_to_bf16(fmaxf(agg[i] + b[d], 0.0f));
}

// ------------------------------------------------------- WMMA bf16 GEMM
// C[N,O] = A[N,K](bf16) @ Bt[O,K](bf16)^T
// Block = 4 waves; wave w owns the 16-row strip tm = blockIdx.x*4 + w and
// accumulates ALL O/16 output tiles, so A streams from HBM exactly once.
// The whole Bt panel (O*K bf16 <= 64KB) is staged into LDS via the
// Tensor Data Mover (tensor_load_to_lds + s_wait_tensorcnt), then B
// fragments come from LDS (ds_load_b128).
template <int K, int O>
__global__ __launch_bounds__(128)
void k_gemm_bf16(const unsigned short* __restrict__ A,
                 const unsigned short* __restrict__ Bt,
                 float* __restrict__ C) {
  constexpr int TILES_O = O / 16;
  __shared__ __align__(128) unsigned short bpanel[O * K];

#if __has_builtin(__builtin_amdgcn_tensor_load_to_lds)
  if (threadIdx.x < 32) {               // wave 0 issues the TDM descriptor
    constexpr unsigned NDW = (unsigned)(O * K / 2);   // panel size in dwords
    unsigned long long ga = (unsigned long long)(__UINTPTR_TYPE__)Bt;
    unsigned lds_off = (unsigned)(__UINTPTR_TYPE__)&bpanel[0]; // low 32b = LDS offset
    u32x4 g0;
    g0[0] = 1u;                                   // count=1, user descriptor
    g0[1] = lds_off;                              // lds_addr
    g0[2] = (unsigned)(ga & 0xFFFFFFFFu);         // global_addr[31:0]
    g0[3] = (unsigned)((ga >> 32) & 0x01FFFFFFu) | (2u << 30);  // ga[56:32] | type=2
    i32x8 g1;
    g1[0] = (int)(2u << 16);          // workgroup_mask=0, data_size=4B
    g1[1] = (int)(NDW << 16);         // tensor_dim0[15:0]   (bits 63:48)
    g1[2] = (int)(1u << 16);          // tensor_dim0 hi=0, tensor_dim1=1
    g1[3] = (int)(NDW << 16);         // tile_dim0 (bits 127:112)
    g1[4] = 1;                        // tile_dim1=1, tile_dim2=0
    g1[5] = (int)NDW;                 // tensor_dim0_stride[31:0]
    g1[6] = (int)(NDW << 16);         // stride hi=0, tensor_dim1_stride[15:0]
    g1[7] = 0;
    i32x4 g2 = {0, 0, 0, 0};
    i32x4 g3 = {0, 0, 0, 0};
#if __clang_major__ >= 23
    i32x8 g4 = {0, 0, 0, 0, 0, 0, 0, 0};
    __builtin_amdgcn_tensor_load_to_lds(g0, g1, g2, g3, g4, 0);
#else
    __builtin_amdgcn_tensor_load_to_lds(g0, g1, g2, g3, 0);
#endif
#if __has_builtin(__builtin_amdgcn_s_wait_tensorcnt)
    __builtin_amdgcn_s_wait_tensorcnt(0);
#else
    asm volatile("s_wait_tensorcnt 0x0" ::: "memory");
#endif
  }
#else
  // fallback: cooperative load of the panel
  for (int i = threadIdx.x * 8; i < O * K; i += 128 * 8)
    *(u16x8*)&bpanel[i] = *(const u16x8*)&Bt[i];
#endif
  __syncthreads();

  int tm   = blockIdx.x * 4 + (threadIdx.x >> 5);   // wave -> 16-row strip
  int lane = threadIdx.x & 31;
  int half = lane >> 4;        // 0 or 1
  int l15  = lane & 15;
  const unsigned short* arow = A + (long)(tm * 16 + l15) * K;

  f32x8 acc[TILES_O] = {};
  for (int k0 = 0; k0 < K; k0 += 32) {
    __builtin_prefetch(arow + k0 + 64, 0, 0);       // global_prefetch_b8
    // A frag: contiguous 16 K-values per lane half (lanes 0-15 K 0-15, lanes 16-31 K 16-31)
    u16x16 araw = *(const u16x16*)(arow + k0 + half * 16);
    bf16x16 av = __builtin_bit_cast(bf16x16, araw);
#pragma unroll
    for (int tn = 0; tn < TILES_O; ++tn) {
      // B frag: col = l15 of tile tn, contiguous 16 K-values per lane half
      const unsigned short* brow =
          bpanel + (tn * 16 + l15) * K + k0 + half * 16;
      u16x16 braw = *(const u16x16*)brow;
      bf16x16 bv = __builtin_bit_cast(bf16x16, braw);
      acc[tn] = __builtin_amdgcn_wmma_f32_16x16x32_bf16(
          false, av, false, bv, (short)0, acc[tn], false, false);
    }
  }
  // C layout: VGPR r -> row tm*16 + half*8 + r, col tn*16 + l15
  long mbase = (long)tm * 16 + half * 8;
#pragma unroll
  for (int tn = 0; tn < TILES_O; ++tn) {
    int n = tn * 16 + l15;
#pragma unroll
    for (int r = 0; r < 8; ++r)
      C[(mbase + r) * O + n] = acc[tn][r];
  }
}

// global mean pool of relu(agg4 + b4) per graph
__global__ void k_pool(const float* __restrict__ agg4, const float* __restrict__ b4,
                       float* seq_t) {
  int i = blockIdx.x * blockDim.x + threadIdx.x;   // NGRAPH*D4 = 1024
  if (i >= NGRAPH * D4) return;
  int g = i / D4;
  int d = i - g * D4;
  float bb = b4[d];
  float acc = 0.0f;
  long base = (long)g * NPGRAPH * D4 + d;
  for (int j = 0; j < NPGRAPH; ++j)
    acc += fmaxf(agg4[base + (long)j * D4] + bb, 0.0f);
  seq_t[i] = acc * (1.0f / (float)NPGRAPH);
}

// LSTM: one block per graph, 128 threads = hidden units, 8 sequential steps
__global__ __launch_bounds__(LH)
void k_lstm(const float* __restrict__ seq, const float* __restrict__ Wih,
            const float* __restrict__ Whh, const float* __restrict__ bih,
            const float* __restrict__ bhh, float* hlast) {
  __shared__ float h[LH], c[LH], xin[D4];
  int g = blockIdx.x;
  int j = threadIdx.x;
  h[j] = 0.0f; c[j] = 0.0f;
  __syncthreads();
  for (int t = 0; t < TSTEPS; ++t) {
    if (j < D4) xin[j] = seq[((long)t * NGRAPH + g) * D4 + j];
    __syncthreads();
    float gates[4];
#pragma unroll
    for (int q = 0; q < 4; ++q) {
      int row = q * LH + j;
      float acc = bih[row] + bhh[row];
      const float* wi = Wih + (long)row * D4;
#pragma unroll 4
      for (int k = 0; k < D4; ++k) acc += xin[k] * wi[k];
      const float* wh = Whh + (long)row * LH;
#pragma unroll 4
      for (int k = 0; k < LH; ++k) acc += h[k] * wh[k];
      gates[q] = acc;
    }
    __syncthreads();   // all reads of h done before overwrite
    float ig = 1.0f / (1.0f + __expf(-gates[0]));
    float fg = 1.0f / (1.0f + __expf(-gates[1]));
    float gg = tanhf(gates[2]);
    float og = 1.0f / (1.0f + __expf(-gates[3]));
    float cn = fg * c[j] + ig * gg;
    c[j] = cn;
    h[j] = og * tanhf(cn);
    __syncthreads();
  }
  hlast[(long)g * LH + j] = h[j];
}

__global__ void k_fc(const float* __restrict__ hlast, const float* __restrict__ Wfc,
                     const float* __restrict__ bfc, float* out) {
  int i = threadIdx.x;
  if (i >= NGRAPH * 2) return;
  int g = i >> 1, o = i & 1;
  float acc = bfc[o];
#pragma unroll 4
  for (int k = 0; k < LH; ++k) acc += hlast[(long)g * LH + k] * Wfc[k * 2 + o];
  out[i] = acc;
}

// ================================================================= host side
static inline unsigned cdiv(long a, long b) { return (unsigned)((a + b - 1) / b); }

extern "C" void kernel_launch(void* const* d_in, const int* in_sizes, int n_in,
                              void* d_out, int out_size, void* d_ws, size_t ws_size,
                              hipStream_t stream) {
  const float* x   = (const float*)d_in[0];
  const int*   ei  = (const int*)d_in[1];
  // d_in[2] = batch (unused: graphs are contiguous, 1080 nodes each)
  const float* W1  = (const float*)d_in[3];
  const float* b1  = (const float*)d_in[4];
  const float* W2  = (const float*)d_in[5];
  const float* b2  = (const float*)d_in[6];
  const float* W3  = (const float*)d_in[7];
  const float* b3  = (const float*)d_in[8];
  const float* W4  = (const float*)d_in[9];
  const float* b4  = (const float*)d_in[10];
  const float* Wih = (const float*)d_in[11];
  const float* Whh = (const float*)d_in[12];
  const float* bih = (const float*)d_in[13];
  const float* bhh = (const float*)d_in[14];
  const float* Wfc = (const float*)d_in[15];
  const float* bfc = (const float*)d_in[16];
  float* out = (float*)d_out;

  // ---- carve workspace (aligned 256B)
  char* p = (char*)d_ws;
  auto carve = [&](size_t bytes) -> char* {
    char* r = p; p += (bytes + 255) & ~(size_t)255; return r;
  };
  float* deg   = (float*)carve((size_t)N_NODES * 4);
  float* dis   = (float*)carve((size_t)N_NODES * 4);
  float* norm  = (float*)carve((size_t)N_TOT * 4);
  float* aggS  = (float*)carve((size_t)N_NODES * DIN * 4);
  unsigned short* actA = (unsigned short*)carve((size_t)N_NODES * D1 * 2);
  unsigned short* actB = (unsigned short*)carve((size_t)N_NODES * D2 * 2);
  float* tbuf   = (float*)carve((size_t)N_NODES * D2 * 4);
  float* aggbuf = (float*)carve((size_t)N_NODES * D2 * 4);
  unsigned short* wt2 = (unsigned short*)carve((size_t)D1 * D2 * 2);
  unsigned short* wt3 = (unsigned short*)carve((size_t)D2 * D3 * 2);
  unsigned short* wt4 = (unsigned short*)carve((size_t)D3 * D4 * 2);
  float* seq   = (float*)carve((size_t)TSTEPS * NGRAPH * D4 * 4);
  float* hlast = (float*)carve((size_t)NGRAPH * LH * 4);

  const int B = 256;

  // ---- GCN normalization (once per launch)
  k_fill_f32<<<cdiv(N_NODES, B), B, 0, stream>>>(deg, N_NODES, 1.0f);
  k_deg <<<cdiv(N_EDGES, B), B, 0, stream>>>(ei, deg);
  k_dis <<<cdiv(N_NODES, B), B, 0, stream>>>(deg, dis);
  k_norm<<<cdiv(N_TOT,   B), B, 0, stream>>>(ei, dis, norm);

  // ---- weights: transpose + convert to bf16 (once per launch)
  k_wt<<<cdiv(D1 * D2, B), B, 0, stream>>>(W2, wt2, D1, D2);
  k_wt<<<cdiv(D2 * D3, B), B, 0, stream>>>(W3, wt3, D2, D3);
  k_wt<<<cdiv(D3 * D4, B), B, 0, stream>>>(W4, wt4, D3, D4);

  const int gemmBlocks = (N_NODES / 16) / 4;   // 4 waves (row strips) per block

  for (int t = 0; t < TSTEPS; ++t) {
    const float* xt = x + (size_t)t * N_NODES * DIN;

    // layer 1: aggregate raw input (3 dims) then transform 3->256 (+relu, ->bf16)
    k_fill_f32<<<cdiv((long)N_NODES * DIN, B), B, 0, stream>>>(aggS, (long)N_NODES * DIN, 0.0f);
    k_edge_agg<<<cdiv((long)N_TOT * DIN, B), B, 0, stream>>>(ei, norm, xt, aggS, DIN);
    k_conv1<<<cdiv((long)N_NODES * D1, B), B, 0, stream>>>(aggS, W1, b1, actA);

    // layer 2: transform FIRST (256->128, WMMA) then aggregate 128 dims
    k_gemm_bf16<D1, D2><<<gemmBlocks, 128, 0, stream>>>(actA, wt2, tbuf);
    k_fill_f32<<<cdiv((long)N_NODES * D2, B), B, 0, stream>>>(aggbuf, (long)N_NODES * D2, 0.0f);
    k_edge_agg<<<cdiv((long)N_TOT * D2, B), B, 0, stream>>>(ei, norm, tbuf, aggbuf, D2);
    k_bias_relu_cvt<<<cdiv((long)N_NODES * D2, B), B, 0, stream>>>(
        aggbuf, b2, actB, D2, (long)N_NODES * D2);

    // layer 3: 128->64 WMMA, aggregate 64 dims
    k_gemm_bf16<D2, D3><<<gemmBlocks, 128, 0, stream>>>(actB, wt3, tbuf);
    k_fill_f32<<<cdiv((long)N_NODES * D3, B), B, 0, stream>>>(aggbuf, (long)N_NODES * D3, 0.0f);
    k_edge_agg<<<cdiv((long)N_TOT * D3, B), B, 0, stream>>>(ei, norm, tbuf, aggbuf, D3);
    k_bias_relu_cvt<<<cdiv((long)N_NODES * D3, B), B, 0, stream>>>(
        aggbuf, b3, actA, D3, (long)N_NODES * D3);

    // layer 4: 64->32 WMMA, aggregate 32 dims, pool (bias+relu fused in pool)
    k_gemm_bf16<D3, D4><<<gemmBlocks, 128, 0, stream>>>(actA, wt4, tbuf);
    k_fill_f32<<<cdiv((long)N_NODES * D4, B), B, 0, stream>>>(aggbuf, (long)N_NODES * D4, 0.0f);
    k_edge_agg<<<cdiv((long)N_TOT * D4, B), B, 0, stream>>>(ei, norm, tbuf, aggbuf, D4);
    k_pool<<<cdiv(NGRAPH * D4, B), B, 0, stream>>>(aggbuf, b4, seq + (size_t)t * NGRAPH * D4);
  }

  // ---- LSTM over the 8-step sequence, then FC
  k_lstm<<<NGRAPH, LH, 0, stream>>>(seq, Wih, Whh, bih, bhh, hlast);
  k_fc<<<1, 64, 0, stream>>>(hlast, Wfc, bfc, out);
}
